// AgentEncoderL2_78288663872337
// MI455X (gfx1250) — matrix operand order repaired and checked by reading, not compiled, not run on hardware
//
#include <hip/hip_runtime.h>
#include <hip/hip_bf16.h>
#include <math.h>
#include <stdint.h>

// ---------------------------------------------------------------------------
// MI455X (gfx1250) fused agent-encoder layer.
// wave32, WMMA 16x16x32 bf16 (fp32 accum). All heavy contractions on v_wmma.
// Key fusion: out_rel = (attn-reduced pf) @ W_rel  -> avoids 603MB intermediate.
// pf tiles staged to LDS via global_load_async_to_lds_b128 (ASYNCcnt path).
// ---------------------------------------------------------------------------

typedef __bf16 bf16;
typedef __attribute__((ext_vector_type(16))) __bf16 v16bf;
typedef __attribute__((ext_vector_type(8)))  float  v8f;

#define BATCH    4
#define SEQ      384
#define D_MODEL  256
#define N_HEADS  8
#define D_HEAD   32
#define D_REL    7
#define MLP_HID  16
#define BN       (BATCH*SEQ)          // 1536
#define QKV_COLS (3*D_MODEL)          // 768
#define LDP      392                  // padded probs row stride (floats)
#define PF_CHUNK 64                   // j-chunk staged per async round
#define PFLD     452                  // padded pf-slab row stride (floats): 452%64=4

// ---- workspace layout (bytes) ----
#define WS_XLN    0u          // bf16 [BN,256]            786432
#define WS_WQKVT  786432u     // bf16 [768,256]           393216
#define WS_WOUTT  1179648u    // bf16 [256,256]           131072
#define WS_Q      1310720u    // bf16 [B,H,N,32] (pre-scaled by 1/sqrt(Dh))
#define WS_K      2097152u    // bf16 [B,H,N,32]
#define WS_VT     2883584u    // bf16 [B,H,32,N] (transposed for B-frag)
#define WS_BIAS   3670016u    // f32  [B,H,N,N]           18874368
#define WS_O      22544384u   // bf16 [BN,256] attention output

// ---------------------------------------------------------------------------
// WMMA fragment helpers (CDNA5 wave32 layouts, CDNA5 ISA §7.12.2)
// ---------------------------------------------------------------------------
static __device__ __forceinline__ v8f wmma_bf16(v16bf a, v16bf b, v8f c) {
  return __builtin_amdgcn_wmma_f32_16x16x32_bf16(
      /*neg_a=*/false, a, /*neg_b=*/false, b,
      /*c_mod=*/(short)0, c, /*reuse_a=*/false, /*reuse_b=*/false);
}

// A (16x32, MxK) from row-major bf16, base at [row0,k0], leading dim ld.
static __device__ __forceinline__ v16bf load_frag_a_bf16(const bf16* base, int ld) {
  const int lane = threadIdx.x & 31;
  const int h = lane >> 4, m = lane & 15;
  const bf16* rp = base + (size_t)m * ld;
  v16bf a;
#pragma unroll
  for (int v = 0; v < 8; ++v) {
    const int k = ((v & 3) << 1) + (h << 3) + ((v >> 2) << 4);
    a[2 * v]     = rp[k];
    a[2 * v + 1] = rp[k + 1];
  }
  return a;
}

// Same A layout but sourced from fp32 (LDS softmax probabilities), cvt->bf16.
static __device__ __forceinline__ v16bf load_frag_a_f32(const float* base, int ld) {
  const int lane = threadIdx.x & 31;
  const int h = lane >> 4, m = lane & 15;
  const float* rp = base + (size_t)m * ld;
  v16bf a;
#pragma unroll
  for (int v = 0; v < 8; ++v) {
    const int k = ((v & 3) << 1) + (h << 3) + ((v >> 2) << 4);
    a[2 * v]     = (bf16)rp[k];
    a[2 * v + 1] = (bf16)rp[k + 1];
  }
  return a;
}

// B (32x16, KxN) from *transposed* row-major Bt[N][K] (Bt[n][k] = B[k][n]).
static __device__ __forceinline__ v16bf load_frag_b_bt(const bf16* base, int ld) {
  const int lane = threadIdx.x & 31;
  const int h = lane >> 4, n = lane & 15;
  const bf16* rp = base + (size_t)n * ld;
  v16bf b;
#pragma unroll
  for (int v = 0; v < 8; ++v) {
    const int k = ((v & 3) << 1) + (h << 3) + ((v >> 2) << 4);
    b[2 * v]     = rp[k];
    b[2 * v + 1] = rp[k + 1];
  }
  return b;
}

// async 16B global -> LDS (ASYNCcnt); lds_off = DS-relative byte address
static __device__ __forceinline__ void async_copy_b128(uint32_t lds_off, const void* gsrc) {
  const uint64_t ga = (uint64_t)(uintptr_t)gsrc;
  asm volatile("global_load_async_to_lds_b128 %0, %1, off"
               :
               : "v"(lds_off), "v"(ga)
               : "memory");
}
static __device__ __forceinline__ void wait_asynccnt0() {
  asm volatile("s_wait_asynccnt 0x0" ::: "memory");
}

// ---------------------------------------------------------------------------
// Kernel 1: convert + transpose weights to bf16 (tiny, once per launch)
// ---------------------------------------------------------------------------
__global__ void prep_weights_kernel(const float* __restrict__ Wqkv,
                                    const float* __restrict__ Wout,
                                    bf16* __restrict__ wqkvT,
                                    bf16* __restrict__ woutT) {
  const int i = blockIdx.x * blockDim.x + threadIdx.x;
  const int n0 = D_MODEL * QKV_COLS;
  const int n1 = D_MODEL * D_MODEL;
  if (i < n0) {
    const int k = i / QKV_COLS, c = i % QKV_COLS;
    wqkvT[(size_t)c * D_MODEL + k] = (bf16)Wqkv[i];
  } else if (i < n0 + n1) {
    const int j = i - n0;
    const int k = j / D_MODEL, c = j % D_MODEL;
    woutT[(size_t)c * D_MODEL + k] = (bf16)Wout[j];
  }
}

// ---------------------------------------------------------------------------
// Kernel 2: LayerNorm, fp32 in -> bf16 out. One 256-thread block per row.
// ---------------------------------------------------------------------------
__global__ void layernorm_kernel(const float* __restrict__ x,
                                 const float* __restrict__ gamma,
                                 const float* __restrict__ beta,
                                 bf16* __restrict__ y) {
  const int row = blockIdx.x;
  const int t = threadIdx.x;
  const float v = x[(size_t)row * D_MODEL + t];
  float s1 = v, s2 = v * v;
#pragma unroll
  for (int off = 16; off >= 1; off >>= 1) {
    s1 += __shfl_xor(s1, off, 32);
    s2 += __shfl_xor(s2, off, 32);
  }
  __shared__ float a1[8], a2[8];
  const int w = t >> 5;
  if ((t & 31) == 0) { a1[w] = s1; a2[w] = s2; }
  __syncthreads();
  float S1 = 0.f, S2 = 0.f;
#pragma unroll
  for (int k = 0; k < 8; ++k) { S1 += a1[k]; S2 += a2[k]; }
  const float mean = S1 * (1.0f / 256.0f);
  const float var  = S2 * (1.0f / 256.0f) - mean * mean;
  const float out  = (v - mean) * rsqrtf(var + 1e-5f) * gamma[t] + beta[t];
  y[(size_t)row * D_MODEL + t] = (bf16)out;
}

// ---------------------------------------------------------------------------
// Kernel 3: distance-bias MLP (1->16 gelu ->8 heads) + padding mask, fp32 out
// ---------------------------------------------------------------------------
__global__ void dist_bias_kernel(const float* __restrict__ pdist,
                                 const unsigned char* __restrict__ pmask,
                                 const float* __restrict__ W1, const float* __restrict__ b1,
                                 const float* __restrict__ W2, const float* __restrict__ b2,
                                 float* __restrict__ db) {
  const int idx = blockIdx.x * blockDim.x + threadIdx.x;
  if (idx >= BATCH * SEQ * SEQ) return;
  const int b = idx / (SEQ * SEQ);
  const int rem = idx % (SEQ * SEQ);
  const int i = rem / SEQ, j = rem % SEQ;
  const float d = pdist[idx];
  float bias[N_HEADS];
#pragma unroll
  for (int c = 0; c < N_HEADS; ++c) bias[c] = b2[c];
#pragma unroll
  for (int k = 0; k < MLP_HID; ++k) {
    const float xh = d * W1[k] + b1[k];
    const float g = 0.5f * xh * (1.0f + erff(xh * 0.70710678118654752f)); // exact gelu
#pragma unroll
    for (int c = 0; c < N_HEADS; ++c) bias[c] += g * W2[k * N_HEADS + c];
  }
  const bool masked = pmask[b * SEQ + i] || pmask[b * SEQ + j];
#pragma unroll
  for (int c = 0; c < N_HEADS; ++c) {
    const float o = masked ? -__builtin_inff() : bias[c];
    db[(((size_t)(b * N_HEADS + c) * SEQ) + i) * SEQ + j] = o;
  }
}

// ---------------------------------------------------------------------------
// Kernel 4: QKV GEMM 1536x768x256 on WMMA; scatter into Q/K (row) and V^T.
// ---------------------------------------------------------------------------
__global__ void qkv_gemm_kernel(const bf16* __restrict__ xln,
                                const bf16* __restrict__ wqkvT,
                                bf16* __restrict__ Qb, bf16* __restrict__ Kb,
                                bf16* __restrict__ Vtb) {
  const int wid = (blockIdx.x * blockDim.x + threadIdx.x) >> 5;
  const int NT = QKV_COLS / 16;  // 48
  const int mt = wid / NT, ct = wid % NT;
  if (mt >= BN / 16) return;
  const int lane = threadIdx.x & 31;
  const int lh = lane >> 4, ln = lane & 15;
  v8f acc = {};
#pragma unroll
  for (int kc = 0; kc < D_MODEL / 32; ++kc) {
    v16bf a = load_frag_a_bf16(xln + (size_t)(mt * 16) * D_MODEL + kc * 32, D_MODEL);
    v16bf b = load_frag_b_bt(wqkvT + (size_t)(ct * 16) * D_MODEL + kc * 32, D_MODEL);
    acc = wmma_bf16(a, b, acc);
  }
  const float qscale = 0.17677669529663687f;  // 1/sqrt(32)
#pragma unroll
  for (int v = 0; v < 8; ++v) {
    const int m = mt * 16 + v + (lh << 3);
    const int c = ct * 16 + ln;
    const int b = m / SEQ, n = m % SEQ;
    const int s = c >> 8, r = c & 255;
    const int hh = r >> 5, d = r & 31;
    const float val = acc[v];
    if (s == 0)
      Qb[((size_t)(b * N_HEADS + hh) * SEQ + n) * D_HEAD + d] = (bf16)(val * qscale);
    else if (s == 1)
      Kb[((size_t)(b * N_HEADS + hh) * SEQ + n) * D_HEAD + d] = (bf16)val;
    else
      Vtb[((size_t)(b * N_HEADS + hh) * D_HEAD + d) * SEQ + n] = (bf16)val;
  }
}

// ---------------------------------------------------------------------------
// Kernel 5: fused attention. Block = 8 waves = 8 heads, per (b, 16-row tile).
//  - QK^T (WMMA) + dist_bias -> logits in LDS (16x392 f32 per head)
//  - softmax in-place, then attn@V (WMMA, A-frags from LDS)
//  - pf slab async-staged to LDS once per block (global_load_async_to_lds_b128),
//    consumed by all 8 head-waves for S[i,r]; folded via W_rel.
// ---------------------------------------------------------------------------
extern __shared__ float smem[];
__global__ void attn_kernel(const bf16* __restrict__ Q, const bf16* __restrict__ K,
                            const bf16* __restrict__ Vt, const float* __restrict__ db,
                            const float* __restrict__ pf, const float* __restrict__ Wrel,
                            bf16* __restrict__ O) {
  const int b  = blockIdx.x / (SEQ / 16);
  const int it = blockIdx.x % (SEQ / 16);
  const int i0 = it * 16;
  const int h  = threadIdx.x >> 5;  // wave == head
  const int lane = threadIdx.x & 31;
  const int lh = lane >> 4, ln = lane & 15;

  float* probs = smem + (size_t)h * 16 * LDP;     // per-head logits/probs
  float* pfs   = smem + (size_t)N_HEADS * 16 * LDP;  // shared pf slab [16][PFLD]

  const bf16* Qh  = Q  + ((size_t)(b * N_HEADS + h) * SEQ + i0) * D_HEAD;
  const bf16* Kh  = K  + ((size_t)(b * N_HEADS + h) * SEQ) * D_HEAD;
  const bf16* Vth = Vt + ((size_t)(b * N_HEADS + h) * D_HEAD) * SEQ;
  const float* dbh = db + ((size_t)(b * N_HEADS + h) * SEQ + i0) * SEQ;

  // Q fragment (Dh=32 == one K-chunk); 1/sqrt(Dh) already folded in.
  const v16bf qa = load_frag_a_bf16(Qh, D_HEAD);

  // ---- logits = QK^T + bias -> LDS ----
  for (int jt = 0; jt < SEQ / 16; ++jt) {
    if (jt + 1 < SEQ / 16)
      __builtin_prefetch(Kh + (size_t)(jt + 1) * 16 * D_HEAD, 0, 3);  // global_prefetch_b8
    v16bf kb = load_frag_b_bt(Kh + (size_t)(jt * 16) * D_HEAD, D_HEAD);
    v8f c = {};
    c = wmma_bf16(qa, kb, c);
    const float* dbp = dbh + jt * 16 + ln;
#pragma unroll
    for (int v = 0; v < 8; ++v) {
      const int m = v + (lh << 3);
      probs[m * LDP + jt * 16 + ln] = c[v] + dbp[(size_t)m * SEQ];
    }
  }
  __syncthreads();

  // ---- softmax: row = lane&15, halves (192 cols) split across lane pairs ----
  {
    float* row = probs + ln * LDP;
    const int j0 = lh * 192;
    float mx = -3.0e38f;
    for (int j = 0; j < 192; ++j) mx = fmaxf(mx, row[j0 + j]);
    mx = fmaxf(mx, __shfl_xor(mx, 16, 32));
    float sum = 0.f;
    for (int j = 0; j < 192; ++j) {
      const float e = __expf(row[j0 + j] - mx);
      row[j0 + j] = e;
      sum += e;
    }
    sum += __shfl_xor(sum, 16, 32);
    const float inv = 1.0f / sum;
    for (int j = 0; j < 192; ++j) row[j0 + j] *= inv;
  }
  __syncthreads();

  // ---- out_std = attn @ V : two 16-col d-tiles, 12 K-chunks of 32 ----
  v8f acc0 = {}, acc1 = {};
  for (int jc = 0; jc < SEQ / 32; ++jc) {
    v16bf pa  = load_frag_a_f32(probs + jc * 32, LDP);
    v16bf vb0 = load_frag_b_bt(Vth + (size_t)0 * SEQ + jc * 32, SEQ);
    v16bf vb1 = load_frag_b_bt(Vth + (size_t)16 * SEQ + jc * 32, SEQ);
    acc0 = wmma_bf16(pa, vb0, acc0);
    acc1 = wmma_bf16(pa, vb1, acc1);
  }

  // ---- S[r] = sum_j p[row,j] * pf[b,i0+row,j,r], pf staged via async DMA ----
  // Per 64-j chunk: 16 rows x 64 j x 7 f32 = 1792 aligned 16B packets,
  // exactly 7 per thread. All 8 head-waves consume the same slab from LDS.
  float s[D_REL];
#pragma unroll
  for (int r = 0; r < D_REL; ++r) s[r] = 0.f;

  for (int j0 = 0; j0 < SEQ; j0 += PF_CHUNK) {
    __syncthreads();  // previous slab fully consumed before overwrite
    {
      const int t = (int)threadIdx.x;
#pragma unroll
      for (int p = 0; p < 7; ++p) {
        const int e = t + p * 256;       // 0..1791 packet id
        const int m = e / 112;           // tile row
        const int q = e % 112;           // 16B packet within row
        const float* gsrc =
            pf + (((size_t)b * SEQ + (i0 + m)) * SEQ + j0) * D_REL + q * 4;
        const uint32_t ldst =
            (uint32_t)(uintptr_t)(pfs + (size_t)m * PFLD) + (uint32_t)(q * 16);
        async_copy_b128(ldst, gsrc);
      }
    }
    wait_asynccnt0();   // s_wait_asynccnt 0 : own ASYNC transfers landed
    __syncthreads();    // all waves' transfers visible to everyone
    // consume: row = lane&15, lane-half splits the 64-j chunk
    const float* prow = probs + (size_t)ln * LDP + j0 + lh * 32;
    const float* pfrow = pfs + (size_t)ln * PFLD + (size_t)(lh * 32) * D_REL;
    for (int j = 0; j < 32; ++j) {
      const float pv = prow[j];
#pragma unroll
      for (int r = 0; r < D_REL; ++r) s[r] += pv * pfrow[j * D_REL + r];
    }
  }
#pragma unroll
  for (int r = 0; r < D_REL; ++r) s[r] += __shfl_xor(s[r], 16, 32);

  // ---- fold out_rel = S @ W_rel[:, h*32 + d] into acc tiles, write O (bf16) ----
  const int dcol0 = h * D_HEAD + ln;
  const int dcol1 = h * D_HEAD + 16 + ln;
#pragma unroll
  for (int v = 0; v < 8; ++v) {
    const int m = v + (lh << 3);
    float add0 = 0.f, add1 = 0.f;
#pragma unroll
    for (int r = 0; r < D_REL; ++r) {
      const float sm = __shfl(s[r], m, 32);  // lane m holds S for row m
      add0 += sm * Wrel[r * D_MODEL + dcol0];
      add1 += sm * Wrel[r * D_MODEL + dcol1];
    }
    const size_t orow = (size_t)(b * SEQ + i0 + m) * D_MODEL;
    O[orow + dcol0] = (bf16)(acc0[v] + add0);
    O[orow + dcol1] = (bf16)(acc1[v] + add1);
  }
}

// ---------------------------------------------------------------------------
// Kernel 6: output projection 1536x256x256 (WMMA) + bias + residual, fp32 out
// ---------------------------------------------------------------------------
__global__ void out_gemm_kernel(const bf16* __restrict__ O, const bf16* __restrict__ woutT,
                                const float* __restrict__ bout,
                                const float* __restrict__ resid,
                                float* __restrict__ out) {
  const int wid = (blockIdx.x * blockDim.x + threadIdx.x) >> 5;
  const int NT = D_MODEL / 16;  // 16
  const int mt = wid / NT, ct = wid % NT;
  if (mt >= BN / 16) return;
  const int lane = threadIdx.x & 31;
  const int lh = lane >> 4, ln = lane & 15;
  v8f acc = {};
#pragma unroll
  for (int kc = 0; kc < D_MODEL / 32; ++kc) {
    v16bf a = load_frag_a_bf16(O + (size_t)(mt * 16) * D_MODEL + kc * 32, D_MODEL);
    v16bf b = load_frag_b_bt(woutT + (size_t)(ct * 16) * D_MODEL + kc * 32, D_MODEL);
    acc = wmma_bf16(a, b, acc);
  }
#pragma unroll
  for (int v = 0; v < 8; ++v) {
    const int m = mt * 16 + v + (lh << 3);
    const int c = ct * 16 + ln;
    const size_t idx = (size_t)m * D_MODEL + c;
    out[idx] = acc[v] + bout[c] + resid[idx];
  }
}

// ---------------------------------------------------------------------------
extern "C" void kernel_launch(void* const* d_in, const int* in_sizes, int n_in,
                              void* d_out, int out_size, void* d_ws, size_t ws_size,
                              hipStream_t stream) {
  const float* agent = (const float*)d_in[0];
  const float* pf    = (const float*)d_in[1];
  const float* pdist = (const float*)d_in[2];
  const unsigned char* pmask = (const unsigned char*)d_in[3];
  const float* Wqkv  = (const float*)d_in[4];
  const float* Wout  = (const float*)d_in[5];
  const float* bout  = (const float*)d_in[6];
  const float* W1    = (const float*)d_in[7];
  const float* b1    = (const float*)d_in[8];
  const float* W2    = (const float*)d_in[9];
  const float* b2    = (const float*)d_in[10];
  const float* Wrel  = (const float*)d_in[11];
  const float* gamma = (const float*)d_in[12];
  const float* beta  = (const float*)d_in[13];

  char* ws = (char*)d_ws;
  bf16*  xln   = (bf16*)(ws + WS_XLN);
  bf16*  wqkvT = (bf16*)(ws + WS_WQKVT);
  bf16*  woutT = (bf16*)(ws + WS_WOUTT);
  bf16*  Qb    = (bf16*)(ws + WS_Q);
  bf16*  Kb    = (bf16*)(ws + WS_K);
  bf16*  Vtb   = (bf16*)(ws + WS_VT);
  float* db    = (float*)(ws + WS_BIAS);
  bf16*  Ob    = (bf16*)(ws + WS_O);

  // 1) weight prep (bf16 + transpose)
  {
    const int total = D_MODEL * QKV_COLS + D_MODEL * D_MODEL;
    prep_weights_kernel<<<(total + 255) / 256, 256, 0, stream>>>(Wqkv, Wout, wqkvT, woutT);
  }
  // 2) layernorm
  layernorm_kernel<<<BN, 256, 0, stream>>>(agent, gamma, beta, xln);
  // 3) distance bias MLP
  {
    const int total = BATCH * SEQ * SEQ;
    dist_bias_kernel<<<(total + 255) / 256, 256, 0, stream>>>(pdist, pmask, W1, b1, W2, b2, db);
  }
  // 4) QKV GEMM: 96*48 tiles, 1 wave each, 8 waves/block
  qkv_gemm_kernel<<<(BN / 16) * (QKV_COLS / 16) / 8, 256, 0, stream>>>(xln, wqkvT, Qb, Kb, Vtb);
  // 5) fused attention: (b, i-tile) blocks, 8 waves = 8 heads,
  //    LDS = 8*16*392*4 (probs) + 16*452*4 (pf slab) = 229,632 B  (<320KB WGP LDS)
  {
    const size_t shmem = (size_t)N_HEADS * 16 * LDP * sizeof(float)
                       + (size_t)16 * PFLD * sizeof(float);
    attn_kernel<<<BATCH * (SEQ / 16), 256, shmem, stream>>>(Qb, Kb, Vtb, db, pf, Wrel, Ob);
  }
  // 6) output projection + bias + residual
  out_gemm_kernel<<<(BN / 16) * (D_MODEL / 16) / 8, 256, 0, stream>>>(
      Ob, woutT, bout, agent, (float*)d_out);
}
